// MPNNPOM_63307817943727
// MI455X (gfx1250) — matrix operand order repaired and checked by reading, not compiled, not run on hardware
//
#include <hip/hip_runtime.h>
#include <hip/hip_bf16.h>

// =====================================================================
// MPNN forward for MI455X (gfx1250, wave32, WMMA).
// Dominant GEMM (E=65536 x K=128 x N=4096) + GRU/LSTM gate GEMMs run on
// v_wmma_f32_16x16x32_f16 (fp16 in, fp32 accum). Each wave computes a
// 32(M) x 64(N) strip: 8 WMMAs per 32-wide K-step, with all fragments
// loaded into distinct registers first so loads overlap WMMA issue.
// Scatter/segment ops use fp32 global atomics.
// Requires ws_size >= ~700 MB (512 MB fp16 W_e).
// =====================================================================

typedef __attribute__((ext_vector_type(16))) _Float16 v16h;
typedef __attribute__((ext_vector_type(8)))  _Float16 v8h;
typedef __attribute__((ext_vector_type(8)))  float    v8f;

__device__ __forceinline__ float sigf(float x) { return 1.0f / (1.0f + __expf(-x)); }

__device__ __forceinline__ v16h ld_frag32(const _Float16* p0, const _Float16* p1) {
    v8h a = *(const v8h*)p0;
    v8h b = *(const v8h*)p1;
    return __builtin_shufflevector(a, b, 0,1,2,3,4,5,6,7,8,9,10,11,12,13,14,15);
}

// ---------------- utility kernels ----------------
__global__ void k_fill_f32(float* p, size_t n, float v) {
    size_t i = (size_t)blockIdx.x * blockDim.x + threadIdx.x;
    if (i < n) p[i] = v;
}
__global__ void k_fill_f16(_Float16* p, size_t n) {
    size_t i = (size_t)blockIdx.x * blockDim.x + threadIdx.x;
    if (i < n) p[i] = (_Float16)0.0f;
}
__global__ void k_f32_to_f16(const float* __restrict__ s, _Float16* __restrict__ d, size_t n) {
    size_t i = (size_t)blockIdx.x * blockDim.x + threadIdx.x;
    if (i < n) d[i] = (_Float16)s[i];
}

// ---------------- WMMA GEMM: C[M,N] = act(A16[M,K] @ W16[N,K]^T + bias (+C)) ---
// K % 32 == 0, M % 128 == 0, N % 64 == 0. 128 threads = 4 waves; each wave
// computes a 32(M) x 64(N) strip (2 A-frags x 4 B-frags = 8 WMMAs / K-step).
template<bool RELU, bool ACCUM, bool OUT16>
__launch_bounds__(128)
__global__ void k_gemm_wmma(const _Float16* __restrict__ A, const _Float16* __restrict__ W,
                            const float* __restrict__ bias, float* __restrict__ C,
                            _Float16* __restrict__ C16, int M, int N, int K) {
    const int lane  = threadIdx.x & 31;
    const int wave  = threadIdx.x >> 5;
    const int l     = lane & 15;
    const bool hi   = lane >= 16;
    const int aoff  = hi ? 16 : 0;   // A-fragment K sub-offset for hi lanes
    const int boff  = hi ? 16 : 0;   // B-fragment K sub-offset for hi lanes
    const int tileM  = blockIdx.y * 128 + wave * 32;
    const int tileN0 = blockIdx.x * 64;

    v8f acc[2][4] = {};
    const _Float16* Arow0 = A + (size_t)(tileM + l) * K + aoff;
    const _Float16* Arow1 = A + (size_t)(tileM + 16 + l) * K + aoff;
    const _Float16* Wrow0 = W + (size_t)(tileN0 + 0 * 16 + l) * K + boff;
    const _Float16* Wrow1 = W + (size_t)(tileN0 + 1 * 16 + l) * K + boff;
    const _Float16* Wrow2 = W + (size_t)(tileN0 + 2 * 16 + l) * K + boff;
    const _Float16* Wrow3 = W + (size_t)(tileN0 + 3 * 16 + l) * K + boff;

    for (int k0 = 0; k0 < K; k0 += 32) {
        // All fragments for this K-step into distinct registers first, so the
        // compiler can retire loads with partial s_wait_loadcnt and overlap
        // VMEM with WMMA issue.
        v16h af0 = ld_frag32(Arow0 + k0, Arow0 + k0 + 8);
        v16h af1 = ld_frag32(Arow1 + k0, Arow1 + k0 + 8);
        v16h bf0 = ld_frag32(Wrow0 + k0, Wrow0 + k0 + 8);
        v16h bf1 = ld_frag32(Wrow1 + k0, Wrow1 + k0 + 8);
        v16h bf2 = ld_frag32(Wrow2 + k0, Wrow2 + k0 + 8);
        v16h bf3 = ld_frag32(Wrow3 + k0, Wrow3 + k0 + 8);
        acc[0][0] = __builtin_amdgcn_wmma_f32_16x16x32_f16(false, af0, false, bf0, (short)0, acc[0][0], false, false);
        acc[0][1] = __builtin_amdgcn_wmma_f32_16x16x32_f16(false, af0, false, bf1, (short)0, acc[0][1], false, false);
        acc[0][2] = __builtin_amdgcn_wmma_f32_16x16x32_f16(false, af0, false, bf2, (short)0, acc[0][2], false, false);
        acc[0][3] = __builtin_amdgcn_wmma_f32_16x16x32_f16(false, af0, false, bf3, (short)0, acc[0][3], false, false);
        acc[1][0] = __builtin_amdgcn_wmma_f32_16x16x32_f16(false, af1, false, bf0, (short)0, acc[1][0], false, false);
        acc[1][1] = __builtin_amdgcn_wmma_f32_16x16x32_f16(false, af1, false, bf1, (short)0, acc[1][1], false, false);
        acc[1][2] = __builtin_amdgcn_wmma_f32_16x16x32_f16(false, af1, false, bf2, (short)0, acc[1][2], false, false);
        acc[1][3] = __builtin_amdgcn_wmma_f32_16x16x32_f16(false, af1, false, bf3, (short)0, acc[1][3], false, false);
    }
#pragma unroll
    for (int rb = 0; rb < 2; ++rb) {
        const int mb = tileM + rb * 16 + (hi ? 8 : 0);
#pragma unroll
        for (int t = 0; t < 4; ++t) {
            const int n = tileN0 + t * 16 + l;
            const float bv = bias ? bias[n] : 0.0f;
#pragma unroll
            for (int r = 0; r < 8; ++r) {
                size_t idx = (size_t)(mb + r) * N + n;
                float v = acc[rb][t][r] + bv;
                if (ACCUM) v += C[idx];
                if (RELU && v < 0.0f) v = 0.0f;
                if (OUT16) C16[idx] = (_Float16)v;
                else       C[idx]   = v;
            }
        }
    }
}

// ---------------- generic VALU dense: out = act(In[M,K] @ W[N,K]^T + b) ------
template<int ACT>  // 0=none 1=relu 2=sigmoid
__global__ void k_dense(const float* __restrict__ In, const float* __restrict__ W,
                        const float* __restrict__ b, float* __restrict__ out32,
                        _Float16* __restrict__ out16, int M, int N, int K) {
    size_t idx = (size_t)blockIdx.x * blockDim.x + threadIdx.x;
    if (idx >= (size_t)M * N) return;
    int m = (int)(idx / N), n = (int)(idx % N);
    const float* in = In + (size_t)m * K;
    const float* w  = W + (size_t)n * K;
    float acc = b ? b[n] : 0.0f;
    for (int k = 0; k < K; ++k) acc += in[k] * w[k];
    if (ACT == 1 && acc < 0.0f) acc = 0.0f;
    if (ACT == 2) acc = sigf(acc);
    if (out32) out32[idx] = acc;
    if (out16) out16[idx] = (_Float16)acc;
}

// ---------------- per-edge einsum + scatter: agg[dst] += h[src] @ W_e --------
__global__ void k_einsum_scatter(const float* __restrict__ h, const _Float16* __restrict__ We,
                                 const int* __restrict__ esrc, const int* __restrict__ edst,
                                 float* __restrict__ agg, int E) {
    const int sub = threadIdx.x >> 6;       // 4 edges per 256-thread block
    const int j   = threadIdx.x & 63;
    const int e   = blockIdx.x * 4 + sub;
    __shared__ float sh[4][64];
    if (e < E) sh[sub][j] = h[(size_t)esrc[e] * 64 + j];
    __syncthreads();
    if (e >= E) return;
    const _Float16* we = We + (size_t)e * 4096;
    float msg = 0.0f;
#pragma unroll 8
    for (int i = 0; i < 64; ++i) msg += sh[sub][i] * (float)we[i * 64 + j];
    atomicAdd(&agg[(size_t)edst[e] * 64 + j], msg);
}

// ---------------- m = relu(agg + h + b) as fp16 (GRU input) ------------------
__global__ void k_gru_pre(const float* __restrict__ agg, const float* __restrict__ h32,
                          const float* __restrict__ nb, _Float16* __restrict__ m16, int N) {
    size_t idx = (size_t)blockIdx.x * blockDim.x + threadIdx.x;
    if (idx >= (size_t)N * 64) return;
    float v = agg[idx] + h32[idx] + nb[idx & 63];
    m16[idx] = (_Float16)(v > 0.0f ? v : 0.0f);
}

// ---------------- GRU gate combine -------------------------------------------
__global__ void k_gru_gate(const float* __restrict__ gi, const float* __restrict__ gh,
                           float* __restrict__ h32, _Float16* __restrict__ h16, int N) {
    size_t idx = (size_t)blockIdx.x * blockDim.x + threadIdx.x;
    if (idx >= (size_t)N * 64) return;
    int n = (int)(idx >> 6), j = (int)(idx & 63);
    const float* gin = gi + (size_t)n * 192;
    const float* ghn = gh + (size_t)n * 192;
    float r  = sigf(gin[j]       + ghn[j]);
    float z  = sigf(gin[64 + j]  + ghn[64 + j]);
    float nn = tanhf(gin[128 + j] + r * ghn[128 + j]);
    float hn = (1.0f - z) * nn + z * h32[idx];
    h32[idx] = hn;
    h16[idx] = (_Float16)hn;
}

// ---------------- readout: x[dst] += concat(h[src], relu(ef@peW^T+peb)) ------
__global__ void k_readout(const float* __restrict__ h32, const float* __restrict__ ef,
                          const float* __restrict__ peW, const float* __restrict__ peb,
                          const int* __restrict__ esrc, const int* __restrict__ edst,
                          float* __restrict__ x, int E) {
    int e = blockIdx.x, t = threadIdx.x;   // 128 threads
    if (e >= E) return;
    float v;
    if (t < 64) {
        v = h32[(size_t)esrc[e] * 64 + t];
    } else {
        int c = t - 64;
        float a = peb[c];
#pragma unroll
        for (int k = 0; k < 6; ++k) a += ef[(size_t)e * 6 + k] * peW[c * 6 + k];
        v = a > 0.0f ? a : 0.0f;
    }
    atomicAdd(&x[(size_t)edst[e] * 128 + t], v);
}

// ---------------- graph node-range discovery (node2graph is sorted) ----------
__global__ void k_range_init(int* gs, int* ge, int B, int N) {
    int i = blockIdx.x * blockDim.x + threadIdx.x;
    if (i < B) { gs[i] = N; ge[i] = 0; }
}
__global__ void k_range_scan(const int* __restrict__ n2g, int* gs, int* ge, int N) {
    int i = blockIdx.x * blockDim.x + threadIdx.x;
    if (i >= N) return;
    int g = n2g[i];
    atomicMin(&gs[g], i);
    atomicMax(&ge[g], i + 1);
}

// ---------------- LSTM gate combine ------------------------------------------
__global__ void k_lstm_gate(const float* __restrict__ g, float* __restrict__ cs,
                            float* __restrict__ hs, _Float16* __restrict__ hs16, int B) {
    size_t idx = (size_t)blockIdx.x * blockDim.x + threadIdx.x;
    if (idx >= (size_t)B * 128) return;
    int b = (int)(idx >> 7), j = (int)(idx & 127);
    const float* gr = g + (size_t)b * 512;
    float ig = sigf(gr[j]);
    float fg = sigf(gr[128 + j]);
    float gg = tanhf(gr[256 + j]);
    float og = sigf(gr[384 + j]);
    float c = fg * cs[idx] + ig * gg;
    float h = og * tanhf(c);
    cs[idx] = c;
    hs[idx] = h;
    hs16[idx] = (_Float16)h;
}

// ---------------- set2set attention (one block / graph, 128 threads) ---------
__global__ void k_attention(const float* __restrict__ x, const float* __restrict__ q,
                            const int* __restrict__ gs, const int* __restrict__ ge,
                            float* __restrict__ ebuf, float* __restrict__ qstar,
                            _Float16* __restrict__ qstar16) {
    int g = blockIdx.x, t = threadIdx.x;     // 128 threads
    __shared__ float sq[128];
    __shared__ float red[128];
    sq[t] = q[(size_t)g * 128 + t];
    __syncthreads();
    int s = gs[g], e = ge[g];
    float mx = -3.4e38f;
    for (int i = s + t; i < e; i += 128) {
        const float* xr = x + (size_t)i * 128;
        float d = 0.0f;
        for (int k = 0; k < 128; ++k) d += xr[k] * sq[k];
        ebuf[i] = d;
        mx = fmaxf(mx, d);
    }
    red[t] = mx; __syncthreads();
    for (int o = 64; o > 0; o >>= 1) { if (t < o) red[t] = fmaxf(red[t], red[t + o]); __syncthreads(); }
    mx = red[0]; __syncthreads();
    float sum = 0.0f;
    for (int i = s + t; i < e; i += 128) {
        float ex = __expf(ebuf[i] - mx);
        ebuf[i] = ex;
        sum += ex;
    }
    red[t] = sum; __syncthreads();
    for (int o = 64; o > 0; o >>= 1) { if (t < o) red[t] += red[t + o]; __syncthreads(); }
    sum = red[0];
    float inv = (sum > 0.0f) ? 1.0f / sum : 0.0f;
    float r = 0.0f;
    for (int i = s; i < e; ++i) r += ebuf[i] * x[(size_t)i * 128 + t];   // coalesced over t
    r *= inv;
    size_t q0 = (size_t)g * 256;
    qstar[q0 + t]         = sq[t];
    qstar[q0 + 128 + t]   = r;
    qstar16[q0 + t]       = (_Float16)sq[t];
    qstar16[q0 + 128 + t] = (_Float16)r;
}

// =====================================================================
// Host orchestration
// =====================================================================
static inline unsigned cdiv(size_t a, size_t b) { return (unsigned)((a + b - 1) / b); }

extern "C" void kernel_launch(void* const* d_in, const int* in_sizes, int n_in,
                              void* d_out, int out_size, void* d_ws, size_t ws_size,
                              hipStream_t stream) {
    // ---- inputs (dict order, params flattened leaf-by-leaf) ----
    const float* node_feats = (const float*)d_in[0];   // [N,134]
    const float* edge_feats = (const float*)d_in[1];   // [E,6]
    const int*   esrc       = (const int*)d_in[2];
    const int*   edst       = (const int*)d_in[3];
    const int*   n2g        = (const int*)d_in[4];
    const float* projW = (const float*)d_in[5],  *projB = (const float*)d_in[6];
    const float* enW1  = (const float*)d_in[7],  *enB1  = (const float*)d_in[8];
    const float* enW2  = (const float*)d_in[9],  *enB2  = (const float*)d_in[10];
    const float* nncB  = (const float*)d_in[11];
    const float* gWih  = (const float*)d_in[12], *gWhh  = (const float*)d_in[13];
    const float* gBih  = (const float*)d_in[14], *gBhh  = (const float*)d_in[15];
    const float* peW   = (const float*)d_in[16], *peB   = (const float*)d_in[17];
    const float* lWih[3], *lWhh[3], *lBih[3], *lBhh[3];
    for (int l = 0; l < 3; ++l) {
        lWih[l] = (const float*)d_in[18 + 4 * l];
        lWhh[l] = (const float*)d_in[19 + 4 * l];
        lBih[l] = (const float*)d_in[20 + 4 * l];
        lBhh[l] = (const float*)d_in[21 + 4 * l];
    }
    const float* fW0 = (const float*)d_in[30], *fB0 = (const float*)d_in[31];
    const float* fW1 = (const float*)d_in[32], *fB1 = (const float*)d_in[33];
    const float* fW2 = (const float*)d_in[34], *fB2 = (const float*)d_in[35];
    float* out = (float*)d_out;

    const int N = in_sizes[4];            // 32768 nodes
    const int E = in_sizes[2];            // 65536 edges
    const int B = out_size / 138;         // 1024 graphs

    // ---- workspace carve (256B aligned) ----
    char* ws = (char*)d_ws;
    size_t off = 0;
    auto carve = [&](size_t bytes) -> void* {
        void* p = ws + off;
        off = (off + bytes + 255) & ~(size_t)255;
        return p;
    };
    _Float16* We16    = (_Float16*)carve((size_t)E * 4096 * 2);   // 512 MB
    _Float16* a16     = (_Float16*)carve((size_t)E * 128 * 2);
    _Float16* enW2_16 = (_Float16*)carve((size_t)4096 * 128 * 2);
    float*    h32     = (float*)carve((size_t)N * 64 * 4);
    _Float16* h16     = (_Float16*)carve((size_t)N * 64 * 2);
    float*    agg     = (float*)carve((size_t)N * 64 * 4);
    _Float16* m16     = (_Float16*)carve((size_t)N * 64 * 2);
    float*    gi      = (float*)carve((size_t)N * 192 * 4);
    float*    gh      = (float*)carve((size_t)N * 192 * 4);
    _Float16* gWih16  = (_Float16*)carve((size_t)192 * 64 * 2);
    _Float16* gWhh16  = (_Float16*)carve((size_t)192 * 64 * 2);
    float*    x32     = (float*)carve((size_t)N * 128 * 4);
    int*      gstart  = (int*)carve((size_t)B * 4);
    int*      gend    = (int*)carve((size_t)B * 4);
    const size_t lwSz[6] = {512*256, 512*128, 512*128, 512*128, 512*128, 512*128};
    _Float16* lw16[6];
    for (int i = 0; i < 6; ++i) lw16[i] = (_Float16*)carve(lwSz[i] * 2);
    float*    qstar   = (float*)carve((size_t)B * 256 * 4);
    _Float16* qstar16 = (_Float16*)carve((size_t)B * 256 * 2);
    float*    hs32    = (float*)carve((size_t)3 * B * 128 * 4);
    _Float16* hs16    = (_Float16*)carve((size_t)3 * B * 128 * 2);
    float*    cs      = (float*)carve((size_t)3 * B * 128 * 4);
    float*    gbuf    = (float*)carve((size_t)B * 512 * 4);
    float*    ebuf    = (float*)carve((size_t)N * 4);
    float*    z1      = (float*)carve((size_t)B * 300 * 4);
    float*    z2      = (float*)carve((size_t)B * 256 * 4);
    (void)ws_size; (void)n_in;

    const dim3 T256(256);
    auto cvt = [&](const float* s, _Float16* d, size_t n) {
        k_f32_to_f16<<<cdiv(n, 256), T256, 0, stream>>>(s, d, n);
    };
    auto zero32 = [&](float* p, size_t n) {
        k_fill_f32<<<cdiv(n, 256), T256, 0, stream>>>(p, n, 0.0f);
    };
    auto zero16 = [&](_Float16* p, size_t n) {
        k_fill_f16<<<cdiv(n, 256), T256, 0, stream>>>(p, n);
    };

    // ---- weight conversions to fp16 ----
    cvt(enW2, enW2_16, (size_t)4096 * 128);
    cvt(gWih, gWih16, (size_t)192 * 64);
    cvt(gWhh, gWhh16, (size_t)192 * 64);
    for (int l = 0; l < 3; ++l) { cvt(lWih[l], lw16[2*l], lwSz[2*l]); cvt(lWhh[l], lw16[2*l+1], lwSz[2*l+1]); }

    // ---- node projection: h = relu(nf @ projW^T + b), fp32 + fp16 ----
    k_dense<1><<<cdiv((size_t)N * 64, 256), T256, 0, stream>>>(
        node_feats, projW, projB, h32, h16, N, 64, 134);

    // ---- edge hidden a = relu(ef @ W1^T + b1), fp16 ----
    k_dense<1><<<cdiv((size_t)E * 128, 256), T256, 0, stream>>>(
        edge_feats, enW1, enB1, nullptr, a16, E, 128, 6);

    // ---- big WMMA GEMM: W_e = a @ W2^T + b2 (fp16 out), M=E,K=128,N=4096 ----
    k_gemm_wmma<false, false, true><<<dim3(4096 / 64, E / 128), dim3(128), 0, stream>>>(
        a16, enW2_16, enB2, nullptr, We16, E, 4096, 128);

    // ---- 3 message-passing steps ----
    for (int step = 0; step < 3; ++step) {
        zero32(agg, (size_t)N * 64);
        k_einsum_scatter<<<cdiv(E, 4), T256, 0, stream>>>(h32, We16, esrc, edst, agg, E);
        k_gru_pre<<<cdiv((size_t)N * 64, 256), T256, 0, stream>>>(agg, h32, nncB, m16, N);
        // gi = m @ Wih^T + bih ; gh = hidden @ Whh^T + bhh (WMMA, K=64, N=192)
        k_gemm_wmma<false, false, false><<<dim3(192 / 64, N / 128), dim3(128), 0, stream>>>(
            m16, gWih16, gBih, gi, nullptr, N, 192, 64);
        k_gemm_wmma<false, false, false><<<dim3(192 / 64, N / 128), dim3(128), 0, stream>>>(
            h16, gWhh16, gBhh, gh, nullptr, N, 192, 64);
        k_gru_gate<<<cdiv((size_t)N * 64, 256), T256, 0, stream>>>(gi, gh, h32, h16, N);
    }

    // ---- readout: x[dst] += concat(h[src], relu(pe(ef))) ----
    zero32(x32, (size_t)N * 128);
    k_readout<<<dim3(E), dim3(128), 0, stream>>>(h32, edge_feats, peW, peB, esrc, edst, x32, E);

    // ---- per-graph node ranges (node2graph sorted) ----
    k_range_init<<<cdiv(B, 256), T256, 0, stream>>>(gstart, gend, B, N);
    k_range_scan<<<cdiv(N, 256), T256, 0, stream>>>(n2g, gstart, gend, N);

    // ---- set2set ----
    zero32(qstar, (size_t)B * 256);
    zero16(qstar16, (size_t)B * 256);
    zero32(hs32, (size_t)3 * B * 128);
    zero16(hs16, (size_t)3 * B * 128);
    zero32(cs, (size_t)3 * B * 128);
    for (int it = 0; it < 6; ++it) {
        const _Float16* lin = qstar16;
        int kin = 256;
        for (int l = 0; l < 3; ++l) {
            // g = layer_in @ Wih^T + bih  (WMMA, M=B, N=512)
            k_gemm_wmma<false, false, false><<<dim3(512 / 64, B / 128), dim3(128), 0, stream>>>(
                lin, lw16[2 * l], lBih[l], gbuf, nullptr, B, 512, kin);
            // g += hs[l] @ Whh^T + bhh  (WMMA accumulate, K=128)
            k_gemm_wmma<false, true, false><<<dim3(512 / 64, B / 128), dim3(128), 0, stream>>>(
                hs16 + (size_t)l * B * 128, lw16[2 * l + 1], lBhh[l], gbuf, nullptr, B, 512, 128);
            k_lstm_gate<<<cdiv((size_t)B * 128, 256), T256, 0, stream>>>(
                gbuf, cs + (size_t)l * B * 128, hs32 + (size_t)l * B * 128,
                hs16 + (size_t)l * B * 128, B);
            lin = hs16 + (size_t)l * B * 128;
            kin = 128;
        }
        // attention with q = hs[2]; writes q_star = concat(q, r)
        k_attention<<<dim3(B), dim3(128), 0, stream>>>(
            x32, hs32 + (size_t)2 * B * 128, gstart, gend, ebuf, qstar, qstar16);
    }

    // ---- FFN 256 -> 300 -> 256 -> 138, sigmoid out ----
    k_dense<1><<<cdiv((size_t)B * 300, 256), T256, 0, stream>>>(qstar, fW0, fB0, z1, nullptr, B, 300, 256);
    k_dense<1><<<cdiv((size_t)B * 256, 256), T256, 0, stream>>>(z1, fW1, fB1, z2, nullptr, B, 256, 300);
    k_dense<2><<<cdiv((size_t)B * 138, 256), T256, 0, stream>>>(z2, fW2, fB2, out, nullptr, B, 138, 256);
}
